// Attention_32813550141753
// MI455X (gfx1250) — matrix operand order repaired and checked by reading, not compiled, not run on hardware
//
#include <hip/hip_runtime.h>
#include <hip/hip_bf16.h>

// ---------------------------------------------------------------------------
// Problem constants (match reference)
// ---------------------------------------------------------------------------
#define BB 2
#define TT 2048
#define DD 4096
#define HH 32
#define GG 8
#define HDIM 128
#define EQKV 6144           // H*HD + 2*G*HD
#define NTOK (BB*TT)        // 4096 tokens
#define SCALE 0.08838834764831845f  // 1/sqrt(128)
#define NEG_BIG (-3.0e38f)

// LDS tile strides (elements), padded to de-conflict the 64 LDS banks
#define KT_LD 136           // 32 x 136 bf16  (272B rows)
#define VT_LD 40            // 128 x 40 bf16  (80B rows)
#define PT_LD 40            // 16 x 40 bf16 per wave

typedef unsigned short u16;
typedef unsigned int   u32;
typedef __attribute__((ext_vector_type(16))) __bf16 v16bf;
typedef __attribute__((ext_vector_type(8)))  float  v8f;
typedef __attribute__((ext_vector_type(4)))  u32    v4u;
typedef __attribute__((ext_vector_type(4)))  int    v4i;

union FragU { v4u u[2]; v16bf v; };

__device__ __forceinline__ u16 f2bf(float f) {
    u32 u = __float_as_uint(f);
    u += 0x7FFFu + ((u >> 16) & 1u);      // round-to-nearest-even
    return (u16)(u >> 16);
}

// ---------------------------------------------------------------------------
// 16-byte staging copy global->LDS.  Uses gfx1250 async-to-LDS instructions
// (ASYNCcnt path) when the toolchain exposes them; otherwise a plain
// load+ds_store pair.
// ---------------------------------------------------------------------------
#if __has_builtin(__builtin_amdgcn_global_load_async_to_lds_b128) && \
    __has_builtin(__builtin_amdgcn_s_wait_asynccnt)
#define HAVE_ASYNC_LDS 1
#else
#define HAVE_ASYNC_LDS 0
#endif

__device__ __forceinline__ void cp16(const u16* g, u16* l) {
#if HAVE_ASYNC_LDS
    __builtin_amdgcn_global_load_async_to_lds_b128(
        (__attribute__((address_space(1))) v4i*)g,
        (__attribute__((address_space(3))) v4i*)l, 0, 0);
#else
    *(v4u*)l = *(const v4u*)g;
#endif
}

__device__ __forceinline__ void cp16_wait() {
#if HAVE_ASYNC_LDS
    __builtin_amdgcn_s_wait_asynccnt(0);
#endif
}

// ---------------------------------------------------------------------------
// WMMA helpers.  CDNA5 16x16x32 bf16 layouts (ISA 7.12.2):
//  A (16x32, MxK): lane m=l&15, kh=l>>4; VGPR0..3 = K[kh*8..+7],
//                  VGPR4..7 = K[16+kh*8..+7]        -> two 16B runs.
//  B (32x16, KxN): lane n=l&15, kh=l>>4; VGPR0..3 = K[kh*8..+7],
//                  VGPR4..7 = K[16+kh*8..+7]        -> two 16B runs.
//  C (16x16 f32):  VGPR r = (M=r, lanes 0-15)/(M=r+8, lanes 16-31), N=l&15.
// ---------------------------------------------------------------------------
__device__ __forceinline__ v16bf load_frag_a(const u16* p0, int ld) {
    const int lane = threadIdx.x & 31;
    const u16* p = p0 + (lane & 15) * ld + ((lane >> 4) << 3);
    FragU f;
    f.u[0] = *(const v4u*)(p);
    f.u[1] = *(const v4u*)(p + 16);
    return f.v;
}

__device__ __forceinline__ v16bf load_frag_b(const u16* p0, int ld) {
    const int lane = threadIdx.x & 31;
    const u16* p = p0 + (lane & 15) * ld + ((lane >> 4) << 3);
    FragU f;
    f.u[0] = *(const v4u*)(p);
    f.u[1] = *(const v4u*)(p + 16);
    return f.v;
}

__device__ __forceinline__ v8f wmma_bf16(v16bf a, v16bf b, v8f c) {
    return __builtin_amdgcn_wmma_f32_16x16x32_bf16(
        false, a, false, b, (short)0, c, false, false);
}

// ---------------------------------------------------------------------------
// fp32 -> bf16 convert
// ---------------------------------------------------------------------------
__global__ __launch_bounds__(256) void cvt_bf16_kernel(
        const float* __restrict__ in, u16* __restrict__ out, int n) {
    int i = blockIdx.x * 256 + threadIdx.x;
    if (i < n) out[i] = f2bf(in[i]);
}

// ---------------------------------------------------------------------------
// GEMM: C[M,N] = A[M,K] * B[N,K]^T  (bf16 in, f32 out, both K-contiguous).
// Block tile 128(M) x 256(N), 8 waves in 2x4 grid, 64x64 per wave:
// 16 WMMAs per 16 b128 fragment loads per K-step.
// ---------------------------------------------------------------------------
__global__ __launch_bounds__(256) void gemm_bf16_nt_kernel(
        const u16* __restrict__ A, const u16* __restrict__ Bm,
        float* __restrict__ C, int M, int N, int K) {
    const int wave = threadIdx.x >> 5;
    const int lane = threadIdx.x & 31;
    const int m0 = blockIdx.y * 128 + (wave >> 2) * 64;
    const int n0 = blockIdx.x * 256 + (wave & 3) * 64;

    v8f acc[4][4] = {};

    for (int k0 = 0; k0 < K; k0 += 32) {
        // speculative prefetch of next K tile toward near caches
        __builtin_prefetch(A + (size_t)m0 * K + k0 + 32, 0, 3);
        __builtin_prefetch(Bm + (size_t)n0 * K + k0 + 32, 0, 3);

        v16bf a[4], b[4];
#pragma unroll
        for (int i = 0; i < 4; ++i)
            a[i] = load_frag_a(A + (size_t)(m0 + 16 * i) * K + k0, K);
#pragma unroll
        for (int j = 0; j < 4; ++j)
            b[j] = load_frag_b(Bm + (size_t)(n0 + 16 * j) * K + k0, K);
#pragma unroll
        for (int i = 0; i < 4; ++i)
#pragma unroll
            for (int j = 0; j < 4; ++j)
                acc[i][j] = wmma_bf16(a[i], b[j], acc[i][j]);
    }

    const int rsel = (lane >> 4) * 8;
    const int ncol = lane & 15;
#pragma unroll
    for (int i = 0; i < 4; ++i)
#pragma unroll
        for (int j = 0; j < 4; ++j) {
            float* out = C + (size_t)(m0 + 16 * i) * N + (n0 + 16 * j);
#pragma unroll
            for (int r = 0; r < 8; ++r)
                out[(size_t)(r + rsel) * N + ncol] = acc[i][j][r];
        }
}

// ---------------------------------------------------------------------------
// RoPE + split/layout: qkv f32 [token][6144] ->
//   qb [b][h][t][hd] (roped), kb [b][g][t][hd] (roped),
//   vt [b][g][hd][t] (transposed so P*V B-fragments are contiguous)
// ---------------------------------------------------------------------------
__global__ __launch_bounds__(256) void rope_split_kernel(
        const float* __restrict__ qkv, u16* __restrict__ qb,
        u16* __restrict__ kb, u16* __restrict__ vt) {
    const int PP = EQKV / 2;
    size_t idx = (size_t)blockIdx.x * 256 + threadIdx.x;
    if (idx >= (size_t)NTOK * PP) return;
    const int token = (int)(idx / PP);
    const int p = (int)(idx % PP);
    const int b = token / TT;
    const int t = token % TT;
    const float* row = qkv + (size_t)token * EQKV;

    if (p < HH * (HDIM / 2)) {                      // ---- Q, roped
        const int h = p >> 6, i = p & 63;
        const float x1 = row[h * HDIM + 2 * i];
        const float x2 = row[h * HDIM + 2 * i + 1];
        const float ang = (float)t * __powf(10000.0f, -(2.0f * i) / (float)HDIM);
        float s, c; __sincosf(ang, &s, &c);
        const size_t o = ((size_t)(b * HH + h) * TT + t) * HDIM + 2 * i;
        qb[o]     = f2bf(x1 * c - x2 * s);
        qb[o + 1] = f2bf(x1 * s + x2 * c);
    } else if (p < (HH + GG) * (HDIM / 2)) {        // ---- K, roped
        const int pk = p - HH * (HDIM / 2);
        const int g = pk >> 6, i = pk & 63;
        const float x1 = row[HH * HDIM + g * HDIM + 2 * i];
        const float x2 = row[HH * HDIM + g * HDIM + 2 * i + 1];
        const float ang = (float)t * __powf(10000.0f, -(2.0f * i) / (float)HDIM);
        float s, c; __sincosf(ang, &s, &c);
        const size_t o = ((size_t)(b * GG + g) * TT + t) * HDIM + 2 * i;
        kb[o]     = f2bf(x1 * c - x2 * s);
        kb[o + 1] = f2bf(x1 * s + x2 * c);
    } else {                                        // ---- V, transposed copy
        const int pv = p - (HH + GG) * (HDIM / 2);
        const int g = pv >> 6, i = pv & 63;
        const float x1 = row[(HH + GG) * HDIM + g * HDIM + 2 * i];
        const float x2 = row[(HH + GG) * HDIM + g * HDIM + 2 * i + 1];
        const size_t base = (size_t)(b * GG + g) * HDIM;
        vt[(base + 2 * i) * TT + t]     = f2bf(x1);
        vt[(base + 2 * i + 1) * TT + t] = f2bf(x2);
    }
}

// ---------------------------------------------------------------------------
// Flash attention. Block = 8 waves = one 128-row Q tile of one (b,h);
// each wave owns a 16-row strip.  Per 32-wide KV tile:
//   * all 256 threads cooperatively stage K (32x128) and V^T (128x32) into
//     padded LDS via async-to-LDS (ASYNCcnt), one copy per WGP instead of 8
//   * S = Q*K^T : 8 wmma from LDS B-fragments
//   * online softmax row stats via 16-lane shfl_xor reductions
//   * P staged via per-wave LDS (C-layout -> A-layout), O += P*V : 8 wmma
// ---------------------------------------------------------------------------
__global__ __launch_bounds__(256) void flash_attn_kernel(
        const u16* __restrict__ qb, const u16* __restrict__ kb,
        const u16* __restrict__ vt, u16* __restrict__ ab,
        const int* __restrict__ wsz_p) {
    const int window = *wsz_p;
    const int QT = TT / 128;
    const int qt = blockIdx.x % QT;
    const int h  = (blockIdx.x / QT) % HH;
    const int b  = blockIdx.x / (QT * HH);
    const int g  = h / (HH / GG);

    const int tid  = threadIdx.x;
    const int wave = tid >> 5;
    const int lane = tid & 31;
    const int qrow0 = qt * 128 + wave * 16;

    const u16* Q   = qb + ((size_t)(b * HH + h) * TT + qrow0) * HDIM;
    const u16* Kp  = kb + (size_t)(b * GG + g) * TT * HDIM;
    const u16* Vtg = vt + (size_t)(b * GG + g) * HDIM * TT;

    __shared__ u16 KtL[32 * KT_LD];          // K tile: [kv][d], padded
    __shared__ u16 VtL[HDIM * VT_LD];        // V tile: [hd][kv], padded
    __shared__ u16 PtL[8 * 16 * PT_LD];      // per-wave P staging
    u16* myp = PtL + wave * (16 * PT_LD);

    v16bf qf[4];
#pragma unroll
    for (int d = 0; d < 4; ++d) qf[d] = load_frag_a(Q + d * 32, HDIM);

    v8f of[8] = {};
    float mrow[8], lrow[8];
#pragma unroll
    for (int r = 0; r < 8; ++r) { mrow[r] = NEG_BIG; lrow[r] = 0.0f; }

    const int rsel = (lane >> 4) * 8;
    const int ncol = lane & 15;
    const int kmax = qt * 128 + 128;         // causal upper bound for block

    for (int kv0 = 0; kv0 < kmax; kv0 += 32) {
        __syncthreads();                     // tiles free to overwrite

        // ---- stage K tile: 32 rows x 128 elems, 16 threads/row, 2 passes
#pragma unroll
        for (int pass = 0; pass < 2; ++pass) {
            const int row = pass * 16 + (tid >> 4);
            const int col = (tid & 15) * 8;
            cp16(Kp + (size_t)(kv0 + row) * HDIM + col,
                 KtL + row * KT_LD + col);
        }
        // ---- stage V^T tile: 128 rows x 32 elems, 4 threads/row, 2 passes
#pragma unroll
        for (int pass = 0; pass < 2; ++pass) {
            const int row = pass * 64 + (tid >> 2);
            const int col = (tid & 3) * 8;
            cp16(Vtg + (size_t)row * TT + kv0 + col,
                 VtL + row * VT_LD + col);
        }
        cp16_wait();
        __syncthreads();                     // tiles visible to all waves

        // ---- S = Q * K^T, two 16x16 C tiles, B-fragments from LDS
        v8f s0 = {}, s1 = {};
#pragma unroll
        for (int d = 0; d < 4; ++d) {
            v16bf b0 = load_frag_b(KtL + d * 32, KT_LD);
            v16bf b1 = load_frag_b(KtL + 16 * KT_LD + d * 32, KT_LD);
            s0 = wmma_bf16(qf[d], b0, s0);
            s1 = wmma_bf16(qf[d], b1, s1);
        }

        // ---- mask + online softmax update
        float alpha[8];
#pragma unroll
        for (int r = 0; r < 8; ++r) {
            const int i  = qrow0 + r + rsel;
            const int j0 = kv0 + ncol;
            const int j1 = kv0 + 16 + ncol;
            float v0 = s0[r] * SCALE, v1 = s1[r] * SCALE;
            if (!((j0 <= i) && (i - j0 < window))) v0 = NEG_BIG;
            if (!((j1 <= i) && (i - j1 < window))) v1 = NEG_BIG;

            float t = fmaxf(v0, v1);          // row max across 16-lane half
            t = fmaxf(t, __shfl_xor(t, 8, 32));
            t = fmaxf(t, __shfl_xor(t, 4, 32));
            t = fmaxf(t, __shfl_xor(t, 2, 32));
            t = fmaxf(t, __shfl_xor(t, 1, 32));

            const float mo = mrow[r];
            const float mn = fmaxf(mo, t);
            const float a  = __expf(mo - mn);
            const float p0 = __expf(v0 - mn);
            const float p1 = __expf(v1 - mn);

            float rs = p0 + p1;               // row sum across half
            rs += __shfl_xor(rs, 8, 32);
            rs += __shfl_xor(rs, 4, 32);
            rs += __shfl_xor(rs, 2, 32);
            rs += __shfl_xor(rs, 1, 32);

            mrow[r]  = mn;
            lrow[r]  = lrow[r] * a + rs;
            alpha[r] = a;

            myp[(r + rsel) * PT_LD + ncol]      = f2bf(p0);
            myp[(r + rsel) * PT_LD + ncol + 16] = f2bf(p1);
        }

        // ---- rescale O, then O += P * V  (V B-fragments from LDS)
#pragma unroll
        for (int n = 0; n < 8; ++n)
#pragma unroll
            for (int r = 0; r < 8; ++r) of[n][r] *= alpha[r];

        v16bf pf = load_frag_a(myp, PT_LD);   // LDS ops in-order per wave
#pragma unroll
        for (int n = 0; n < 8; ++n) {
            v16bf vf = load_frag_b(VtL + (n * 16) * VT_LD, VT_LD);
            of[n] = wmma_bf16(pf, vf, of[n]);
        }
    }

    // ---- normalize & store bf16 [b][t][h*HD+hd]
    float inv[8];
#pragma unroll
    for (int r = 0; r < 8; ++r) inv[r] = 1.0f / lrow[r];
#pragma unroll
    for (int n = 0; n < 8; ++n)
#pragma unroll
        for (int r = 0; r < 8; ++r) {
            const int token = qrow0 + r + rsel;
            ab[((size_t)(b * TT + token) * HH + h) * HDIM + n * 16 + ncol] =
                f2bf(of[n][r] * inv[r]);
        }
}

// ---------------------------------------------------------------------------
// Host-side launcher
// ---------------------------------------------------------------------------
extern "C" void kernel_launch(void* const* d_in, const int* in_sizes, int n_in,
                              void* d_out, int out_size, void* d_ws, size_t ws_size,
                              hipStream_t stream) {
    const float* x     = (const float*)d_in[0];
    const float* w_qkv = (const float*)d_in[1];
    const float* w_o   = (const float*)d_in[2];
    const int*   wsz   = (const int*)d_in[3];
    float* out = (float*)d_out;

    char* ws = (char*)d_ws;
    size_t off = 0;
    u16* xb   = (u16*)(ws + off); off += (size_t)NTOK * DD * 2;
    u16* wqb  = (u16*)(ws + off); off += (size_t)EQKV * DD * 2;
    u16* wob  = (u16*)(ws + off); off += (size_t)DD * DD * 2;
    float* qkvf = (float*)(ws + off); off += (size_t)NTOK * EQKV * 4;
    u16* qbuf = (u16*)(ws + off); off += (size_t)BB * HH * TT * HDIM * 2;
    u16* kbuf = (u16*)(ws + off); off += (size_t)BB * GG * TT * HDIM * 2;
    u16* vbuf = (u16*)(ws + off); off += (size_t)BB * GG * TT * HDIM * 2;
    u16* abuf = (u16*)qkvf;   // qkv f32 dead after RoPE; alias its space

    // 1. fp32 -> bf16 casts
    {
        int n1 = NTOK * DD, n2 = EQKV * DD, n3 = DD * DD;
        cvt_bf16_kernel<<<(n1 + 255) / 256, 256, 0, stream>>>(x, xb, n1);
        cvt_bf16_kernel<<<(n2 + 255) / 256, 256, 0, stream>>>(w_qkv, wqb, n2);
        cvt_bf16_kernel<<<(n3 + 255) / 256, 256, 0, stream>>>(w_o, wob, n3);
    }

    // 2. QKV projection: [4096,6144] = x[4096,4096] * w_qkv[6144,4096]^T
    gemm_bf16_nt_kernel<<<dim3(EQKV / 256, NTOK / 128), 256, 0, stream>>>(
        xb, wqb, qkvf, NTOK, EQKV, DD);

    // 3. RoPE + head split + V transpose
    {
        size_t n = (size_t)NTOK * (EQKV / 2);
        rope_split_kernel<<<(unsigned)((n + 255) / 256), 256, 0, stream>>>(
            qkvf, qbuf, kbuf, vbuf);
    }

    // 4. Flash attention (causal, windowed)
    flash_attn_kernel<<<BB * HH * (TT / 128), 256, 0, stream>>>(
        qbuf, kbuf, vbuf, abuf, wsz);

    // 5. Output projection: out[4096,4096] = attn[4096,4096] * w_o[4096,4096]^T
    gemm_bf16_nt_kernel<<<dim3(DD / 256, NTOK / 128), 256, 0, stream>>>(
        abuf, wob, out, NTOK, DD, DD);
}
